// SequenceAutoencoder_49795850830256
// MI455X (gfx1250) — matrix operand order repaired and checked by reading, not compiled, not run on hardware
//
#include <hip/hip_runtime.h>

// Sequence autoencoder: 4 x LSTM(H=64) over T=256, B=2048, fused bottleneck
// and output projection. All GEMMs via v_wmma_f32_16x16x32_bf16 (wave32).
//
// 16 batch rows per block, 128 blocks, 128 threads (4 waves). Wave j owns
// gate columns [16j,16j+16) of all four gates, so the cell state c stays in
// registers. Weight B-operands (4 K-chunks x 4 gates) are loaded ONCE into
// registers in the exact WMMA B-layout (128 VGPRs/wave) -- the recurrent
// inner loop touches LDS only for the [x_t | h_{t-1}] A-operands and the h
// writeback. Activations use the gfx1250 hardware v_tanh_f32 (sigmoid =
// 0.5*tanh(0.5x)+0.5), keeping the 256-step sequential critical path short.

#define B_    2048
#define T_    256
#define H_    64
#define G_    256   // 4*H
#define F_    40
#define BT    16    // batch rows per block
#define NTH   128   // 4 waves

typedef __bf16 bf16_t;
typedef __attribute__((ext_vector_type(16))) __bf16 v16bf;
typedef __attribute__((ext_vector_type(8)))  float  v8f;

union V16U { uint4 u[2]; v16bf v; };

enum { IN_F32SEQ = 0, IN_BF16SEQ = 1, IN_CONST = 2 };
enum { OUT_SEQ = 0, OUT_FINAL = 1, OUT_PROJ = 2 };

__device__ __forceinline__ float tanh_fast(float x) {
#if __has_builtin(__builtin_amdgcn_tanhf)
  return __builtin_amdgcn_tanhf(x);            // v_tanh_f32 (gfx1250)
#elif __has_builtin(__builtin_amdgcn_tanh_f32)
  return __builtin_amdgcn_tanh_f32(x);
#else
  float t = __expf(-2.0f * fabsf(x));
  float r = (1.0f - t) * __builtin_amdgcn_rcpf(1.0f + t);
  return copysignf(r, x);
#endif
}
__device__ __forceinline__ float sigmf(float x) {
  return fmaf(0.5f, tanh_fast(0.5f * x), 0.5f);
}

// Two 16B LDS loads -> one 16-element bf16 WMMA operand.
__device__ __forceinline__ v16bf lds_pair(const bf16_t* p0, const bf16_t* p1) {
  V16U t;
  t.u[0] = *(const uint4*)p0;
  t.u[1] = *(const uint4*)p1;
  return t.v;
}

template <int IN_MODE, int OUT_MODE>
__global__ __launch_bounds__(NTH, 1) void lstm_layer(
    const float* __restrict__ xf,   // IN_F32SEQ: [B][T][Din]
    const bf16_t* __restrict__ xb,  // IN_BF16SEQ: [B][T][64]; IN_CONST: [B][64]
    const float* __restrict__ Wih, const int Din,   // [256][Din]
    const float* __restrict__ Whh,                  // [256][64]
    const float* __restrict__ bih, const float* __restrict__ bhh,
    bf16_t* __restrict__ yseq,      // OUT_SEQ: [B][T][64] bf16
    float* __restrict__ hTout,      // OUT_FINAL: [B][64] f32
    const float* __restrict__ oW, const float* __restrict__ ob,
    float* __restrict__ oseq)       // OUT_PROJ: [B][T][40] f32
{
  __shared__ __align__(16) bf16_t sX[BT][64];    // input slice   2 KB
  __shared__ __align__(16) bf16_t sH[BT][64];    // hidden state  2 KB

  const int tid  = threadIdx.x;
  const int lane = tid & 31;
  const int wv   = tid >> 5;         // 0..3
  const int b0   = blockIdx.x * BT;

  const int aRow = lane & 15;               // A-matrix M row
  const int aOff = (lane & 16) ? 16 : 0;    // A K sub-chunk
  const int bOff = (lane & 16) ? 16 : 0;    // B K sub-chunk
  const int hi8  = (lane & 16) ? 8 : 0;     // C-layout row offset

  // ---- load weight B-operands once, directly into WMMA B-layout regs ----
  // Combined K axis: [0,64) = Wih (Din cols zero-padded), [64,128) = Whh.
  v16bf Breg[4][4];                          // [kc][gate] : 128 VGPRs
#pragma unroll
  for (int kc = 0; kc < 4; ++kc) {
#pragma unroll
    for (int g = 0; g < 4; ++g) {
      const int ncol = 16 * (wv + 4 * g) + (lane & 15);
      const int kb = 32 * kc + bOff;
      v16bf bv;
#pragma unroll
      for (int i = 0; i < 16; ++i) {
        const int k = kb + i;
        float v;
        if (kc < 2) v = (k < Din) ? Wih[ncol * Din + k] : 0.0f;
        else        v = Whh[ncol * 64 + (k - 64)];
        bv[i] = (bf16_t)v;
      }
      Breg[kc][g] = bv;
    }
  }
  v16bf oBreg[2];                            // out-proj weights (waves 0..2)
  if constexpr (OUT_MODE == OUT_PROJ) {
#pragma unroll
    for (int kc = 0; kc < 2; ++kc) {
      const int ncol = 16 * wv + (lane & 15);   // padded 40 -> 48
      const int kb = 32 * kc + bOff;
      v16bf bv;
#pragma unroll
      for (int i = 0; i < 16; ++i)
        bv[i] = (wv < 3 && ncol < F_) ? (bf16_t)oW[ncol * H_ + kb + i]
                                      : (bf16_t)0.0f;
      oBreg[kc] = bv;
    }
  }

  // ---- zero h, stage first input slice ----
  {
    int r = tid >> 3, q = tid & 7;          // 16 rows x 8 col-chunks
    for (int i = 0; i < 8; ++i) sH[r][q * 8 + i] = (bf16_t)0.0f;
    if constexpr (IN_MODE == IN_CONST) {
      for (int i = 0; i < 8; ++i)
        sX[r][q * 8 + i] = xb[(size_t)(b0 + r) * 64 + q * 8 + i];
    } else if constexpr (IN_MODE == IN_BF16SEQ) {
      *(uint4*)&sX[r][q * 8] =
          *(const uint4*)(xb + (size_t)(b0 + r) * T_ * 64 + q * 8);
    } else {
      for (int i = 0; i < 8; ++i) {
        int c = q * 8 + i;
        float v = (c < Din) ? xf[(size_t)(b0 + r) * T_ * Din + c] : 0.0f;
        sX[r][c] = (bf16_t)v;
      }
    }
  }
  __syncthreads();

  // per-lane gate bias (WMMA C initializer)
  float bc[4];
#pragma unroll
  for (int g = 0; g < 4; ++g) {
    int col = 16 * (wv + 4 * g) + (lane & 15);
    bc[g] = bih[col] + bhh[col];
  }
  float opb = 0.0f;
  if constexpr (OUT_MODE == OUT_PROJ) {
    int c = 16 * wv + (lane & 15);
    opb = (wv < 3 && c < F_) ? ob[c] : 0.0f;
  }

  // cell state lives in registers (C-layout of this wave's 16-col slice)
  v8f cst;
#pragma unroll
  for (int e = 0; e < 8; ++e) cst[e] = 0.0f;

  for (int t = 0; t < T_; ++t) {
    // ---- prefetch x_{t+1} into registers ----
    const int r = tid >> 3, q = tid & 7;
    uint4 xr;
    float xrf[8];
    const bool havex = (IN_MODE != IN_CONST) && (t + 1 < T_);
    if constexpr (IN_MODE == IN_BF16SEQ) {
      if (havex)
        xr = *(const uint4*)(xb + ((size_t)(b0 + r) * T_ + (t + 1)) * 64 + q * 8);
    } else if constexpr (IN_MODE == IN_F32SEQ) {
      if (havex) {
        for (int i = 0; i < 8; ++i) {
          int c = q * 8 + i;
          xrf[i] = (c < Din) ? xf[((size_t)(b0 + r) * T_ + (t + 1)) * Din + c]
                             : 0.0f;
        }
      }
    }

    // ---- gates = [x_t | h_{t-1}] @ [Wih|Whh]^T + b  (K = 128) ----
    v8f acc[4];
#pragma unroll
    for (int g = 0; g < 4; ++g)
#pragma unroll
      for (int e = 0; e < 8; ++e) acc[g][e] = bc[g];

#pragma unroll
    for (int kc = 0; kc < 4; ++kc) {
      const bf16_t* src = (kc < 2) ? &sX[0][0] : &sH[0][0];
      const int kk = (kc & 1) * 32;
      const bf16_t* abase = src + aRow * 64 + kk + aOff;
      v16bf A = lds_pair(abase, abase + 8);
#pragma unroll
      for (int g = 0; g < 4; ++g) {
        acc[g] = __builtin_amdgcn_wmma_f32_16x16x32_bf16(
            false, A, false, Breg[kc][g], (short)0, acc[g], false, false);
      }
    }

    // ---- LSTM pointwise: c,h update entirely in registers ----
    float hv[8];
#pragma unroll
    for (int e = 0; e < 8; ++e) {
      float ig = sigmf(acc[0][e]);
      float fg = sigmf(acc[1][e]);
      float gg = tanh_fast(acc[2][e]);
      float og = sigmf(acc[3][e]);
      float cn = fg * cst[e] + ig * gg;
      cst[e] = cn;
      hv[e] = og * tanh_fast(cn);
    }

    __syncthreads();  // all LDS reads for step t done before overwriting

    // write h_t (bf16) into LDS per C-matrix layout
    {
      const int col = 16 * wv + (lane & 15);
#pragma unroll
      for (int e = 0; e < 8; ++e) sH[e + hi8][col] = (bf16_t)hv[e];
    }
    // commit prefetched x_{t+1}
    if constexpr (IN_MODE == IN_BF16SEQ) {
      if (havex) *(uint4*)&sX[r][q * 8] = xr;
    } else if constexpr (IN_MODE == IN_F32SEQ) {
      if (havex)
        for (int i = 0; i < 8; ++i) sX[r][q * 8 + i] = (bf16_t)xrf[i];
    }

    __syncthreads();  // h_t visible to all waves

    if constexpr (OUT_MODE == OUT_SEQ) {
      // cooperative bf16 store of h_t
      bf16_t* dst = yseq + ((size_t)(b0 + r) * T_ + t) * 64 + q * 8;
      *(uint4*)dst = *(const uint4*)&sH[r][q * 8];
    } else if constexpr (OUT_MODE == OUT_PROJ) {
      // fused out_t = h_t @ out_W^T + out_b  (N padded 40 -> 48, 3 tiles)
      if (wv < 3) {
        v8f oc;
#pragma unroll
        for (int e = 0; e < 8; ++e) oc[e] = opb;
#pragma unroll
        for (int kc = 0; kc < 2; ++kc) {
          const bf16_t* abase = &sH[aRow][32 * kc + aOff];
          v16bf A = lds_pair(abase, abase + 8);
          oc = __builtin_amdgcn_wmma_f32_16x16x32_bf16(
              false, A, false, oBreg[kc], (short)0, oc, false, false);
        }
        const int col = 16 * wv + (lane & 15);
        if (col < F_) {
#pragma unroll
          for (int e = 0; e < 8; ++e)
            oseq[((size_t)(b0 + e + hi8) * T_ + t) * F_ + col] = oc[e];
        }
      }
    }
  }  // t

  if constexpr (OUT_MODE == OUT_FINAL) {
    int r = tid >> 3, q = tid & 7;
    for (int i = 0; i < 8; ++i)
      hTout[(size_t)(b0 + r) * 64 + q * 8 + i] = (float)sH[r][q * 8 + i];
  }
}

// z = hT @ bottle_W^T + bottle_b ; hexp = relu(z @ expand_W^T + expand_b)
__global__ __launch_bounds__(NTH) void bottleneck_expand(
    const float* __restrict__ hT, const float* __restrict__ bW,
    const float* __restrict__ bb, const float* __restrict__ eW,
    const float* __restrict__ eb, float* __restrict__ z,
    bf16_t* __restrict__ hexp)
{
  int b = blockIdx.x * blockDim.x + threadIdx.x;
  if (b >= B_) return;
  float h[64];
#pragma unroll
  for (int k = 0; k < 64; ++k) h[k] = hT[(size_t)b * 64 + k];
  float zv[32];
  for (int j = 0; j < 32; ++j) {
    float s = bb[j];
#pragma unroll
    for (int k = 0; k < 64; ++k) s += bW[j * 64 + k] * h[k];
    zv[j] = s;
    z[(size_t)b * 32 + j] = s;
  }
  for (int k = 0; k < 64; ++k) {
    float s = eb[k];
#pragma unroll
    for (int j = 0; j < 32; ++j) s += eW[k * 32 + j] * zv[j];
    hexp[(size_t)b * 64 + k] = (bf16_t)fmaxf(s, 0.0f);
  }
}

extern "C" void kernel_launch(void* const* d_in, const int* in_sizes, int n_in,
                              void* d_out, int out_size, void* d_ws,
                              size_t ws_size, hipStream_t stream)
{
  (void)in_sizes; (void)n_in; (void)out_size; (void)ws_size;
  const float* x     = (const float*)d_in[0];
  const float* eWih0 = (const float*)d_in[1];
  const float* eWhh0 = (const float*)d_in[2];
  const float* ebih0 = (const float*)d_in[3];
  const float* ebhh0 = (const float*)d_in[4];
  const float* eWih1 = (const float*)d_in[5];
  const float* eWhh1 = (const float*)d_in[6];
  const float* ebih1 = (const float*)d_in[7];
  const float* ebhh1 = (const float*)d_in[8];
  const float* bW    = (const float*)d_in[9];
  const float* bb    = (const float*)d_in[10];
  const float* xW    = (const float*)d_in[11];
  const float* xbias = (const float*)d_in[12];
  const float* dWih0 = (const float*)d_in[13];
  const float* dWhh0 = (const float*)d_in[14];
  const float* dbih0 = (const float*)d_in[15];
  const float* dbhh0 = (const float*)d_in[16];
  const float* dWih1 = (const float*)d_in[17];
  const float* dWhh1 = (const float*)d_in[18];
  const float* dbih1 = (const float*)d_in[19];
  const float* dbhh1 = (const float*)d_in[20];
  const float* oW    = (const float*)d_in[21];
  const float* obias = (const float*)d_in[22];

  // workspace: seq buffer (y0, reused for d0) + hT + hexp
  char* ws = (char*)d_ws;
  bf16_t* seq  = (bf16_t*)ws;  // B*T*64 bf16 = 64 MB
  float*  hT   = (float*)(ws + (size_t)B_ * T_ * H_ * sizeof(bf16_t));
  bf16_t* hexp = (bf16_t*)((char*)hT + (size_t)B_ * H_ * sizeof(float));

  float* out = (float*)d_out;                  // [B][T][40]
  float* z   = out + (size_t)B_ * T_ * F_;     // [B][32]

  dim3 grid(B_ / BT), blk(NTH);

  // encoder layer 0: f32 seq in (Din=40, padded) -> bf16 seq out
  lstm_layer<IN_F32SEQ, OUT_SEQ><<<grid, blk, 0, stream>>>(
      x, nullptr, eWih0, 40, eWhh0, ebih0, ebhh0, seq, nullptr, nullptr,
      nullptr, nullptr);
  // encoder layer 1: bf16 seq in -> final hT only
  lstm_layer<IN_BF16SEQ, OUT_FINAL><<<grid, blk, 0, stream>>>(
      nullptr, seq, eWih1, 64, eWhh1, ebih1, ebhh1, nullptr, hT, nullptr,
      nullptr, nullptr);
  // bottleneck + expand (tiny)
  bottleneck_expand<<<dim3(B_ / NTH), blk, 0, stream>>>(hT, bW, bb, xW, xbias,
                                                        z, hexp);
  // decoder layer 0: constant (broadcast) input -> bf16 seq out (reuse seq)
  lstm_layer<IN_CONST, OUT_SEQ><<<grid, blk, 0, stream>>>(
      nullptr, hexp, dWih0, 64, dWhh0, dbih0, dbhh0, seq, nullptr, nullptr,
      nullptr, nullptr);
  // decoder layer 1: bf16 seq in -> fused output projection
  lstm_layer<IN_BF16SEQ, OUT_PROJ><<<grid, blk, 0, stream>>>(
      nullptr, seq, dWih1, 64, dWhh1, dbih1, dbhh1, nullptr, nullptr, oW,
      obias, out);
}
